// Attention_14989435863047
// MI455X (gfx1250) — matrix operand order repaired and checked by reading, not compiled, not run on hardware
//
#include <hip/hip_runtime.h>
#include <hip/hip_bf16.h>

#define SEQ   4096
#define BATCH 4
#define DM    64
#define NELEM (BATCH * SEQ * DM)   // 1048576 elements per tensor

typedef __bf16 bf16;
typedef __attribute__((ext_vector_type(16))) __bf16 v16bf;
typedef __attribute__((ext_vector_type(8)))  __bf16 v8bf;
typedef __attribute__((ext_vector_type(8)))  float  v8f;
typedef __attribute__((ext_vector_type(4)))  unsigned int v4u;
typedef __attribute__((ext_vector_type(8)))  int v8i;
typedef __attribute__((ext_vector_type(4)))  int v4i;

union V16U { v16bf v; v8bf h[2]; };

__device__ __forceinline__ v8f bf16_mma(v16bf a, v16bf b, v8f c) {
    // D = A(16x32 bf16) * B(32x16 bf16) + C(16x16 f32)
    return __builtin_amdgcn_wmma_f32_16x16x32_bf16(
        /*neg_a=*/false, a, /*neg_b=*/false, b,
        /*c_mod=*/(short)0, c, /*reuse_a=*/false, /*reuse_b=*/false);
}

// ---- A fragment (16x32, bf16 source, row-major) ----
__device__ __forceinline__ v16bf load_a_bf16(const bf16* p, int stride, int m0,
                                             int d0, int lane) {
    int lo = lane & 15, hi = lane >> 4;
    const bf16* row = p + (long)(m0 + lo) * stride;
    V16U u;
    u.h[0] = *(const v8bf*)(row + d0 + hi * 8);
    u.h[1] = *(const v8bf*)(row + d0 + 16 + hi * 8);
    return u.v;
}

// ---- A fragment from fp32 source (convert to bf16) ----
__device__ __forceinline__ v16bf load_a_f32(const float* p, int stride, int m0,
                                            int d0, int lane) {
    int lo = lane & 15, hi = lane >> 4;
    const float* row = p + (long)(m0 + lo) * stride;
    V16U u;
#pragma unroll
    for (int i = 0; i < 8; ++i) {
        u.h[0][i] = (__bf16)row[d0 + hi * 8 + i];
        u.h[1][i] = (__bf16)row[d0 + 16 + hi * 8 + i];
    }
    return u.v;
}

// ---- B fragment (32x16): B[k][n] = src[n0+n][k0+k], src row-major ----
// Works for both global and LDS source pointers (ds_load_b128 when shared).
__device__ __forceinline__ v16bf load_b_bf16(const bf16* src, long stride, int n0,
                                             int k0, int lane) {
    int lo = lane & 15, hi = lane >> 4;
    const bf16* row = src + (long)(n0 + lo) * stride;
    V16U u;
    u.h[0] = *(const v8bf*)(row + k0 + hi * 8);
    u.h[1] = *(const v8bf*)(row + k0 + 16 + hi * 8);
    return u.v;
}

__device__ __forceinline__ v16bf load_b_f32(const float* src, int stride, int n0,
                                            int k0, int lane) {
    int lo = lane & 15, hi = lane >> 4;
    const float* row = src + (long)(n0 + lo) * stride;
    V16U u;
#pragma unroll
    for (int i = 0; i < 8; ++i) {
        u.h[0][i] = (__bf16)row[k0 + hi * 8 + i];
        u.h[1][i] = (__bf16)row[k0 + 16 + hi * 8 + i];
    }
    return u.v;
}

// --------------------------------------------------------------------------
// Tensor Data Mover: 2D tile (global -> LDS), bf16 elements (data_size=2B).
// D# per CDNA5 ISA. 1D tiles: tile_d1 = 0.
// --------------------------------------------------------------------------
__device__ __forceinline__ void tdm_load_2d(unsigned lds_off, const void* gptr,
                                            unsigned tensor_d0, unsigned tensor_d1,
                                            unsigned tile_d0, unsigned tile_d1,
                                            unsigned stride0) {
    unsigned long long ga = (unsigned long long)gptr;
    v4u g0;
    g0[0] = 1u;                                            // count=1 (valid D#)
    g0[1] = lds_off;                                       // lds_addr (bytes)
    g0[2] = (unsigned)(ga & 0xFFFFFFFFu);                  // global_addr[31:0]
    g0[3] = (unsigned)((ga >> 32) & 0x1FFFFFFu) | (2u << 30); // addr[56:32]|type=2
    v8i g1;
    g1[0] = (int)(1u << 16);                               // data_size=1 (2 bytes)
    g1[1] = (int)((tensor_d0 & 0xFFFFu) << 16);            // tensor_dim0[15:0]
    g1[2] = (int)(((tensor_d0 >> 16) & 0xFFFFu) |
                  ((tensor_d1 & 0xFFFFu) << 16));          // td0[31:16] | td1[15:0]
    g1[3] = (int)(((tensor_d1 >> 16) & 0xFFFFu) |
                  ((tile_d0 & 0xFFFFu) << 16));            // td1[31:16] | tile_d0
    g1[4] = (int)(tile_d1 & 0xFFFFu);                      // tile_d1 | tile_d2=0
    g1[5] = (int)stride0;                                  // tensor_dim0_stride
    g1[6] = 0;
    g1[7] = 0;
    v4i z4 = {0, 0, 0, 0};
#if __clang_major__ >= 23
    v8i z8 = {0, 0, 0, 0, 0, 0, 0, 0};
    __builtin_amdgcn_tensor_load_to_lds(g0, g1, z4, z4, z8, 0);
#else
    __builtin_amdgcn_tensor_load_to_lds(g0, g1, z4, z4, 0);
#endif
}

__device__ __forceinline__ unsigned lds_offset(const void* p) {
    // flat shared address: low 32 bits are the LDS byte offset (aperture map)
    return (unsigned)(unsigned long long)p;
}

// --------------------------------------------------------------------------
// Kernel 1: Q/K/V projections. X[B*S,64] @ W^T + b.
// Q,K stored row-major bf16 [B*S,64]; V stored transposed bf16 [B][64][S].
// --------------------------------------------------------------------------
__device__ __forceinline__ void proj_tile(const float* X, const float* W,
                                          const float* bias, int m0, int n0,
                                          int lane, float out[8]) {
    v8f c = {};
#pragma unroll
    for (int k0 = 0; k0 < DM; k0 += 32) {
        v16bf a = load_a_f32(X, DM, m0, k0, lane);
        v16bf b = load_b_f32(W, DM, n0, k0, lane);  // W[n][k] row-major
        c = bf16_mma(a, b, c);
    }
    float bv = bias[n0 + (lane & 15)];
#pragma unroll
    for (int r = 0; r < 8; ++r) out[r] = c[r] + bv;
}

__global__ __launch_bounds__(128) void qkv_proj_kernel(
    const float* __restrict__ q, const float* __restrict__ k,
    const float* __restrict__ v, const float* __restrict__ Wq,
    const float* __restrict__ bq, const float* __restrict__ Wk,
    const float* __restrict__ bk, const float* __restrict__ Wv,
    const float* __restrict__ bv, bf16* __restrict__ Qb,
    bf16* __restrict__ Kb, bf16* __restrict__ Vt) {
    int lane = threadIdx.x & 31;
    int wid  = threadIdx.x >> 5;
    int gw   = blockIdx.x * 4 + wid;   // 0..4095
    int m0   = (gw >> 2) * 16;         // global row in [0, B*S)
    int n0   = (gw & 3) * 16;          // output column tile
    int lo = lane & 15, hi = lane >> 4;
    int batch = m0 / SEQ;
    int sInB  = m0 % SEQ;

    float t[8];
    proj_tile(q, Wq, bq, m0, n0, lane, t);
#pragma unroll
    for (int r = 0; r < 8; ++r)
        Qb[(long)(m0 + r + 8 * hi) * DM + n0 + lo] = (__bf16)t[r];
    proj_tile(k, Wk, bk, m0, n0, lane, t);
#pragma unroll
    for (int r = 0; r < 8; ++r)
        Kb[(long)(m0 + r + 8 * hi) * DM + n0 + lo] = (__bf16)t[r];
    proj_tile(v, Wv, bv, m0, n0, lane, t);
#pragma unroll
    for (int r = 0; r < 8; ++r)
        Vt[(long)batch * DM * SEQ + (long)(n0 + lo) * SEQ + sInB + r + 8 * hi] =
            (__bf16)t[r];
}

// --------------------------------------------------------------------------
// Kernel 2: flash attention. Mask: -inf where i > j (attend to j >= i).
// One block = 64 queries (4 waves x 16 rows). K/V tiles staged per block
// into LDS by the Tensor Data Mover, shared by all 4 waves.
// --------------------------------------------------------------------------
__global__ __launch_bounds__(128) void flash_attn_kernel(
    const bf16* __restrict__ Qb, const bf16* __restrict__ Kb,
    const bf16* __restrict__ Vt, bf16* __restrict__ Ob) {
    __shared__ __align__(16) __bf16 Ktile[32 * DM];   // [key][d]   4KB
    __shared__ __align__(16) __bf16 Vtile[DM * 32];   // [d][key]   4KB
    __shared__ __align__(16) __bf16 pbuf[4][16 * 32]; // per-wave P staging 4KB

    int lane = threadIdx.x & 31;
    int wid  = threadIdx.x >> 5;
    int batch   = blockIdx.x >> 6;            // 64 query-blocks per batch
    int m0block = (blockIdx.x & 63) * 64;     // block's query start in sequence
    int m0s     = m0block + wid * 16;         // this wave's query tile
    int lo = lane & 15, hi = lane >> 4;

    const bf16* Qp = Qb + (long)batch * SEQ * DM;
    const bf16* Kp = Kb + (long)batch * SEQ * DM;
    const bf16* Vp = Vt + (long)batch * DM * SEQ;

    // Q A-fragments, loaded once (D=64 -> two k-chunks)
    v16bf aq0 = load_a_bf16(Qp, DM, m0s, 0, lane);
    v16bf aq1 = load_a_bf16(Qp, DM, m0s, 32, lane);

    v8f o0 = {}, o1 = {}, o2 = {}, o3 = {};
    float mrun[8], lrun[8];
#pragma unroll
    for (int r = 0; r < 8; ++r) { mrun[r] = -1e30f; lrun[r] = 0.0f; }

    __bf16* pl = pbuf[wid];
    const float scale = 0.125f;  // 1/sqrt(64)

    // allowed keys are j >= i; whole block shares one key range
    for (int n0 = m0block; n0 < SEQ; n0 += 32) {
        // ---- TDM: stage K (32x64, contiguous) and V^T (64 rows x 32 keys) ----
        if (wid == 0) {
            tdm_load_2d(lds_offset(Ktile), Kp + (long)n0 * DM,
                        /*tensor_d0=*/32 * DM, /*tensor_d1=*/1,
                        /*tile_d0=*/32 * DM, /*tile_d1=*/0,
                        /*stride0=*/32 * DM);
            __builtin_amdgcn_s_wait_tensorcnt(0);
        } else if (wid == 1) {
            tdm_load_2d(lds_offset(Vtile), Vp + n0,
                        /*tensor_d0=*/SEQ, /*tensor_d1=*/DM,
                        /*tile_d0=*/32, /*tile_d1=*/DM,
                        /*stride0=*/SEQ);
            __builtin_amdgcn_s_wait_tensorcnt(0);
        }
        __syncthreads();

        // ---- scores: two 16x16 tiles over 32 keys (B frags from LDS) ----
        v8f c0 = {}, c1 = {};
        {
            v16bf b = load_b_bf16(Ktile, DM, 0, 0, lane);   c0 = bf16_mma(aq0, b, c0);
            b       = load_b_bf16(Ktile, DM, 0, 32, lane);  c0 = bf16_mma(aq1, b, c0);
            b       = load_b_bf16(Ktile, DM, 16, 0, lane);  c1 = bf16_mma(aq0, b, c1);
            b       = load_b_bf16(Ktile, DM, 16, 32, lane); c1 = bf16_mma(aq1, b, c1);
        }
        // ---- scale + mask + online softmax (row = r + 8*hi, col = lane%16) ----
#pragma unroll
        for (int r = 0; r < 8; ++r) {
            int i = m0s + r + 8 * hi;
            bool k0m = i > (n0 + lo);
            bool k1m = i > (n0 + 16 + lo);
            float s0 = k0m ? -1e30f : c0[r] * scale;
            float s1 = k1m ? -1e30f : c1[r] * scale;
            float mx = fmaxf(s0, s1);
#pragma unroll
            for (int off = 8; off; off >>= 1) mx = fmaxf(mx, __shfl_xor(mx, off, 32));
            float mnew  = fmaxf(mrun[r], mx);
            float alpha = __expf(mrun[r] - mnew);
            // masked entries are exactly 0 (handles fully-masked tiles too)
            float p0 = k0m ? 0.0f : __expf(s0 - mnew);
            float p1 = k1m ? 0.0f : __expf(s1 - mnew);
            float rs = p0 + p1;
#pragma unroll
            for (int off = 8; off; off >>= 1) rs += __shfl_xor(rs, off, 32);
            lrun[r] = lrun[r] * alpha + rs;
            mrun[r] = mnew;
            o0[r] *= alpha; o1[r] *= alpha; o2[r] *= alpha; o3[r] *= alpha;
            // stage P in LDS: row-major [16][32]
            pl[(r + 8 * hi) * 32 + lo]      = (__bf16)p0;
            pl[(r + 8 * hi) * 32 + 16 + lo] = (__bf16)p1;
        }
        asm volatile("s_wait_dscnt 0" ::: "memory");  // per-wave LDS store->load

        // ---- P as A fragment (16x32) ----
        V16U up;
        up.h[0] = *(const v8bf*)(pl + lo * 32 + hi * 8);
        up.h[1] = *(const v8bf*)(pl + lo * 32 + 16 + hi * 8);
        v16bf pa = up.v;

        // ---- P @ V: B fragments from LDS V^T tile (rows = d, cols = key) ----
        v16bf bvf = load_b_bf16(Vtile, 32, 0, 0, lane);  o0 = bf16_mma(pa, bvf, o0);
        bvf       = load_b_bf16(Vtile, 32, 16, 0, lane); o1 = bf16_mma(pa, bvf, o1);
        bvf       = load_b_bf16(Vtile, 32, 32, 0, lane); o2 = bf16_mma(pa, bvf, o2);
        bvf       = load_b_bf16(Vtile, 32, 48, 0, lane); o3 = bf16_mma(pa, bvf, o3);

        __syncthreads();  // all waves done with K/V tiles before next TDM fill
    }

    // ---- normalize and store O (bf16, row-major [B*S, 64]) ----
    bf16* Op = Ob + (long)batch * SEQ * DM;
#pragma unroll
    for (int r = 0; r < 8; ++r) {
        float inv = 1.0f / lrun[r];
        long row = m0s + r + 8 * hi;
        Op[row * DM + 0 + lo]  = (__bf16)(o0[r] * inv);
        Op[row * DM + 16 + lo] = (__bf16)(o1[r] * inv);
        Op[row * DM + 32 + lo] = (__bf16)(o2[r] * inv);
        Op[row * DM + 48 + lo] = (__bf16)(o3[r] * inv);
    }
}

// --------------------------------------------------------------------------
// Kernel 3: output projection. out = O @ Wp^T + bp (fp32 output).
// --------------------------------------------------------------------------
__global__ __launch_bounds__(128) void out_proj_kernel(
    const bf16* __restrict__ Ob, const float* __restrict__ Wp,
    const float* __restrict__ bp, float* __restrict__ out) {
    int lane = threadIdx.x & 31;
    int wid  = threadIdx.x >> 5;
    int gw   = blockIdx.x * 4 + wid;
    int m0   = (gw >> 2) * 16;
    int n0   = (gw & 3) * 16;
    int lo = lane & 15, hi = lane >> 4;

    v8f c = {};
#pragma unroll
    for (int k0 = 0; k0 < DM; k0 += 32) {
        v16bf a = load_a_bf16(Ob, DM, m0, k0, lane);
        v16bf b = load_b_f32(Wp, DM, n0, k0, lane);
        c = bf16_mma(a, b, c);
    }
    float bv = bp[n0 + lo];
#pragma unroll
    for (int r = 0; r < 8; ++r)
        out[(long)(m0 + r + 8 * hi) * DM + n0 + lo] = c[r] + bv;
}

// --------------------------------------------------------------------------
extern "C" void kernel_launch(void* const* d_in, const int* in_sizes, int n_in,
                              void* d_out, int out_size, void* d_ws, size_t ws_size,
                              hipStream_t stream) {
    const float* q  = (const float*)d_in[0];
    const float* k  = (const float*)d_in[1];
    const float* v  = (const float*)d_in[2];
    const float* Wq = (const float*)d_in[3];
    const float* bq = (const float*)d_in[4];
    const float* Wk = (const float*)d_in[5];
    const float* bk = (const float*)d_in[6];
    const float* Wv = (const float*)d_in[7];
    const float* bv = (const float*)d_in[8];
    const float* Wp = (const float*)d_in[9];
    const float* bp = (const float*)d_in[10];
    float* out = (float*)d_out;

    bf16* ws = (bf16*)d_ws;
    bf16* Qb = ws + 0L * NELEM;   // [B*S, 64] bf16
    bf16* Kb = ws + 1L * NELEM;   // [B*S, 64] bf16
    bf16* Vt = ws + 2L * NELEM;   // [B][64][S] bf16 (transposed)
    bf16* Ob = ws + 3L * NELEM;   // [B*S, 64] bf16

    qkv_proj_kernel<<<1024, 128, 0, stream>>>(q, k, v, Wq, bq, Wk, bk, Wv, bv,
                                              Qb, Kb, Vt);
    // 256 query blocks of 64 rows; 4 waves per block share TDM-staged K/V
    flash_attn_kernel<<<256, 128, 0, stream>>>(Qb, Kb, Vt, Ob);
    out_proj_kernel<<<1024, 128, 0, stream>>>(Ob, Wp, bp, out);
}
